// GPT_Layer_42425686950142
// MI455X (gfx1250) — matrix operand order repaired and checked by reading, not compiled, not run on hardware
//
#include <hip/hip_runtime.h>
#include <hip/hip_bf16.h>

// ---------------------------------------------------------------------------
// GPT layer on MI455X (gfx1250, wave32).
// Compute-bound (~100 GFLOP vs ~40MB traffic) -> all GEMMs + attention matmuls
// use v_wmma_f32_16x16x32_bf16 (K=32/instr), fp32 accumulate. Scores never hit
// HBM (flash attention, online softmax). GEMM tiles are staged into LDS by the
// Tensor Data Mover (tensor_load_to_lds, TENSORcnt-tracked, double-buffered)
// when the builtin exists; manual global->LDS staging otherwise.
// ---------------------------------------------------------------------------

typedef __attribute__((ext_vector_type(16))) __bf16 bf16x16;
typedef __attribute__((ext_vector_type(8)))  __bf16 bf16x8;
typedef __attribute__((ext_vector_type(8)))  float  f32x8;
typedef __attribute__((ext_vector_type(4)))  unsigned int u32x4;
typedef __attribute__((ext_vector_type(8)))  int i32x8;
typedef __attribute__((ext_vector_type(4)))  int i32x4;

static constexpr int BB  = 2;
static constexpr int SS  = 2048;
static constexpr int DD  = 768;
static constexpr int HH  = 12;
static constexpr int DKk = 64;
static constexpr int HID = 3072;
static constexpr int MM  = BB * SS;   // 4096 rows

#if defined(__HIP_DEVICE_COMPILE__) && __has_builtin(__builtin_amdgcn_tensor_load_to_lds) && __has_builtin(__builtin_amdgcn_s_wait_tensorcnt)
#define USE_TDM 1
#else
#define USE_TDM 0
#endif

__device__ __forceinline__ f32x8 zero8() {
  f32x8 z;
#pragma unroll
  for (int i = 0; i < 8; ++i) z[i] = 0.0f;
  return z;
}

__device__ __forceinline__ f32x8 wmma_bf16(bf16x16 a, bf16x16 b, f32x8 c) {
  // emits v_wmma_f32_16x16x32_bf16
  return __builtin_amdgcn_wmma_f32_16x16x32_bf16(false, a, false, b, (short)0, c,
                                                 false, false);
}

union FragU { bf16x16 v; bf16x8 h[2]; };

// A fragment (16x32, row-major, leading dim ld in elements).
// lane: m = lane&15, kb = (lane>>4)*8; element i -> k = (i>>3)*16 + kb + (i&7)
__device__ __forceinline__ bf16x16 load_frag_a(const __bf16* p, int ld) {
  int lane = threadIdx.x & 31;
  const __bf16* q = p + (lane & 15) * ld + ((lane >> 4) * 8);
  FragU u;
  u.h[0] = *(const bf16x8*)(q);
  u.h[1] = *(const bf16x8*)(q + 16);
  return u.v;
}

// B fragment (32x16) from transposed storage Bt[n][k], leading dim ld.
// lane: n = lane&15, kb = (lane>>4)*8; element i -> k = (i>>3)*16 + kb + (i&7)
__device__ __forceinline__ bf16x16 load_frag_b(const __bf16* p, int ld) {
  int lane = threadIdx.x & 31;
  const __bf16* q = p + (lane & 15) * ld + ((lane >> 4) * 8);
  FragU u;
  u.h[0] = *(const bf16x8*)(q);
  u.h[1] = *(const bf16x8*)(q + 16);
  return u.v;
}

__device__ __forceinline__ float gelu_f(float x) {
  return 0.5f * x * (1.0f + erff(x * 0.70710678118654752f));
}

#if USE_TDM
// 2D TDM tile load: tile (tile_d1 rows x tile_d0 bf16) from a row-major
// tensor (stride0 elements/row) into LDS at lds_addr, inserting 4 DWORDs of
// pad after every 32 DWORDs (=> 144B LDS row stride for 64-elem rows).
// D# packing per CDNA5 ISA (count=1, type=2, 2B data).
__device__ __forceinline__ void tdm_load_2d(unsigned lds_addr,
                                            unsigned long long gaddr,
                                            unsigned tensor_d0, unsigned tensor_d1,
                                            unsigned long long stride0,
                                            unsigned tile_d0, unsigned tile_d1) {
  u32x4 g0;
  g0[0] = 1u;                                            // count=1, user mode
  g0[1] = lds_addr;                                      // lds_addr (bytes)
  g0[2] = (unsigned)(gaddr & 0xffffffffu);               // global_addr[31:0]
  g0[3] = (unsigned)((gaddr >> 32) & 0x01ffffffu) | 0x80000000u; // [56:32]|type=2
  i32x8 g1;
  g1[0] = (int)((1u << 16) | (1u << 20) | (4u << 22) | (3u << 25));
  //             data=2B     pad_en       every 32DW    pad 4DW
  g1[1] = (int)((tensor_d0 & 0xffffu) << 16);            // tensor_dim0[15:0]
  g1[2] = (int)((tensor_d0 >> 16) | ((tensor_d1 & 0xffffu) << 16));
  g1[3] = (int)((tensor_d1 >> 16) | (tile_d0 << 16));    // dim1 hi | tile_dim0
  g1[4] = (int)(tile_d1 & 0xffffu);                      // tile_dim1 (tile_dim2=0)
  g1[5] = (int)(stride0 & 0xffffffffu);                  // dim0_stride[31:0]
  g1[6] = (int)((stride0 >> 32) & 0xffffu);              // dim0_stride[47:32]
  g1[7] = 0;
  i32x4 z4 = {0, 0, 0, 0};
#if __clang_major__ >= 23
  i32x8 z8 = {0, 0, 0, 0, 0, 0, 0, 0};
  __builtin_amdgcn_tensor_load_to_lds(g0, g1, z4, z4, z8, 0);
#else
  __builtin_amdgcn_tensor_load_to_lds(g0, g1, z4, z4, 0);
#endif
}
#endif

// ---------------------------------------------------------------------------
// LayerNorm (fp32 in) -> bf16 out.  One 256-thread block per row, D=768.
// ---------------------------------------------------------------------------
__global__ void ln_bf16(const float* __restrict__ x, const float* __restrict__ g,
                        const float* __restrict__ b, __bf16* __restrict__ out,
                        int D) {
  int row = blockIdx.x;
  int tid = threadIdx.x;
  const float* xr = x + (long)row * D;
  float s = 0.f, s2 = 0.f;
  for (int i = tid; i < D; i += 256) { float v = xr[i]; s += v; s2 += v * v; }
  __shared__ float rs[256], rs2[256];
  rs[tid] = s; rs2[tid] = s2;
  __syncthreads();
  for (int off = 128; off > 0; off >>= 1) {
    if (tid < off) { rs[tid] += rs[tid + off]; rs2[tid] += rs2[tid + off]; }
    __syncthreads();
  }
  float mu   = rs[0] / D;
  float var  = rs2[0] / D - mu * mu;
  float rstd = rsqrtf(var + 1e-5f);
  for (int i = tid; i < D; i += 256)
    out[(long)row * D + i] = (__bf16)((xr[i] - mu) * rstd * g[i] + b[i]);
}

// ---------------------------------------------------------------------------
// Tiled transpose + fp32->bf16:  src[K][N] -> dst[N][K]  (per z batch).
// ---------------------------------------------------------------------------
__global__ void transpose_cvt(const float* __restrict__ src, __bf16* __restrict__ dst,
                              int K, int N) {
  __shared__ float t[64][65];
  int k0 = blockIdx.x * 64, n0 = blockIdx.y * 64;
  long zo = (long)blockIdx.z * K * N;
  for (int i = threadIdx.x; i < 4096; i += 256) {
    int r = i >> 6, c = i & 63;
    t[r][c] = src[zo + (long)(k0 + r) * N + n0 + c];
  }
  __syncthreads();
  for (int i = threadIdx.x; i < 4096; i += 256) {
    int r = i >> 6, c = i & 63;
    dst[zo + (long)(n0 + r) * K + k0 + c] = (__bf16)t[c][r];
  }
}

// ---------------------------------------------------------------------------
// bf16 WMMA GEMM:  C[M,N] = epilogue(A[M,K] * Bt[N,K]^T)
// Block tile 128x64, 256 threads = 8 waves in a 4x2 grid; each wave owns a
// 32x32 register tile (4 C-frags) -> 8 WMMA per 64-deep K stage, 2 ds_load_b128
// per WMMA. LDS rows padded +8 bf16 (144B stride: 16B-aligned, conflict-free).
// TDM path: wave 0 issues tensor_load_to_lds for A+B of stage s+1 into the
// other LDS buffer, s_wait_tensorcnt(2) retires stage s, barrier, compute.
// Epilogue: optional bias (per-n), exact GELU, fp32 residual; outputs:
//   Cf fp32 row-major | Cb bf16 row-major | Cbt bf16 transposed (ldc = M-dim)
// ---------------------------------------------------------------------------
__global__ void gemm_tc(const __bf16* __restrict__ A, long sA, int lda,
                        const __bf16* __restrict__ Bt, long sB, int ldb,
                        float* Cf, __bf16* Cb, __bf16* Cbt, long sC, int ldc,
                        const float* __restrict__ bias,
                        const float* __restrict__ res, int ldr,
                        int K, int act) {
  __shared__ __bf16 lA[2][128 * 72];
  __shared__ __bf16 lB[2][64 * 72];
  const int tid = threadIdx.x;
  const int w = tid >> 5, lane = tid & 31;
  const int wr = w >> 1;            // wave row 0..3 (32 m-rows each)
  const int wc = w & 1;             // wave col 0..1 (32 n-cols each)
  const int m0 = blockIdx.y * 128, n0 = blockIdx.x * 64;
  const __bf16* Az = A + (long)blockIdx.z * sA;
  const __bf16* Bz = Bt + (long)blockIdx.z * sB;

  f32x8 f[2][2];
#pragma unroll
  for (int ms = 0; ms < 2; ++ms)
#pragma unroll
    for (int ns = 0; ns < 2; ++ns) f[ms][ns] = zero8();

  const int nst = K / 64;

#if USE_TDM
  const unsigned long long gA = (unsigned long long)(uintptr_t)Az +
                                ((unsigned long long)m0 * lda) * 2ull;
  const unsigned long long gB = (unsigned long long)(uintptr_t)Bz +
                                ((unsigned long long)n0 * ldb) * 2ull;
  if (w == 0) {
    tdm_load_2d((unsigned)(uintptr_t)&lA[0][0], gA, (unsigned)K, 128u,
                (unsigned long long)lda, 64u, 128u);
    tdm_load_2d((unsigned)(uintptr_t)&lB[0][0], gB, (unsigned)K, 64u,
                (unsigned long long)ldb, 64u, 64u);
  }
  for (int s = 0; s < nst; ++s) {
    if (w == 0) {
      if (s + 1 < nst) {
        unsigned long long ko = (unsigned long long)(s + 1) * 64ull * 2ull;
        tdm_load_2d((unsigned)(uintptr_t)&lA[(s + 1) & 1][0], gA + ko,
                    (unsigned)K, 128u, (unsigned long long)lda, 64u, 128u);
        tdm_load_2d((unsigned)(uintptr_t)&lB[(s + 1) & 1][0], gB + ko,
                    (unsigned)K, 64u, (unsigned long long)ldb, 64u, 64u);
        __builtin_amdgcn_s_wait_tensorcnt(2);  // stage s (older 2 ops) done
      } else {
        __builtin_amdgcn_s_wait_tensorcnt(0);
      }
    }
    __syncthreads();
    const __bf16* bufA = &lA[s & 1][0];
    const __bf16* bufB = &lB[s & 1][0];
#pragma unroll
    for (int ks = 0; ks < 2; ++ks) {
      bf16x16 a0 = load_frag_a(bufA + (wr * 32 + 0)  * 72 + ks * 32, 72);
      bf16x16 a1 = load_frag_a(bufA + (wr * 32 + 16) * 72 + ks * 32, 72);
      bf16x16 b0 = load_frag_b(bufB + (wc * 32 + 0)  * 72 + ks * 32, 72);
      bf16x16 b1 = load_frag_b(bufB + (wc * 32 + 16) * 72 + ks * 32, 72);
      f[0][0] = wmma_bf16(a0, b0, f[0][0]);
      f[0][1] = wmma_bf16(a0, b1, f[0][1]);
      f[1][0] = wmma_bf16(a1, b0, f[1][0]);
      f[1][1] = wmma_bf16(a1, b1, f[1][1]);
    }
    __syncthreads();
  }
#else
  for (int s = 0; s < nst; ++s) {
    int k0 = s * 64;
#pragma unroll
    for (int it = 0; it < 4; ++it) {  // A: 128 rows x 8 chunks
      int idx = tid + it * 256;
      int r = idx >> 3, c = (idx & 7) * 8;
      *(bf16x8*)(&lA[0][0] + r * 72 + c) =
          *(const bf16x8*)(Az + (long)(m0 + r) * lda + k0 + c);
    }
#pragma unroll
    for (int it = 0; it < 2; ++it) {  // B: 64 rows x 8 chunks
      int idx = tid + it * 256;
      int r = idx >> 3, c = (idx & 7) * 8;
      *(bf16x8*)(&lB[0][0] + r * 72 + c) =
          *(const bf16x8*)(Bz + (long)(n0 + r) * ldb + k0 + c);
    }
    if (k0 + 64 < K) {  // near-cache prefetch of next stage
      int r = tid >> 1, c = (tid & 1) * 32;
      __builtin_prefetch(Az + (long)(m0 + r) * lda + k0 + 64 + c, 0, 3);
      if (tid < 128) {
        int r2 = tid >> 1;
        __builtin_prefetch(Bz + (long)(n0 + r2) * ldb + k0 + 64 + c, 0, 3);
      }
    }
    __syncthreads();
#pragma unroll
    for (int ks = 0; ks < 2; ++ks) {
      bf16x16 a0 = load_frag_a(&lA[0][0] + (wr * 32 + 0)  * 72 + ks * 32, 72);
      bf16x16 a1 = load_frag_a(&lA[0][0] + (wr * 32 + 16) * 72 + ks * 32, 72);
      bf16x16 b0 = load_frag_b(&lB[0][0] + (wc * 32 + 0)  * 72 + ks * 32, 72);
      bf16x16 b1 = load_frag_b(&lB[0][0] + (wc * 32 + 16) * 72 + ks * 32, 72);
      f[0][0] = wmma_bf16(a0, b0, f[0][0]);
      f[0][1] = wmma_bf16(a0, b1, f[0][1]);
      f[1][0] = wmma_bf16(a1, b0, f[1][0]);
      f[1][1] = wmma_bf16(a1, b1, f[1][1]);
    }
    __syncthreads();
  }
#endif

  // epilogue
  float* Cfz = Cf ? Cf + (long)blockIdx.z * sC : nullptr;
  __bf16* Cbz = Cb ? Cb + (long)blockIdx.z * sC : nullptr;
  __bf16* Cbtz = Cbt ? Cbt + (long)blockIdx.z * sC : nullptr;
  const int hi = lane >> 4, nl = lane & 15;
#pragma unroll
  for (int ms = 0; ms < 2; ++ms) {
#pragma unroll
    for (int ns = 0; ns < 2; ++ns) {
      f32x8 acc = f[ms][ns];
      int n = n0 + wc * 32 + ns * 16 + nl;
      float bv = bias ? bias[n] : 0.0f;
      float vv[8];
#pragma unroll
      for (int r = 0; r < 8; ++r) {
        int m = m0 + wr * 32 + ms * 16 + hi * 8 + r;
        float v = acc[r] + bv;
        if (act == 1) v = gelu_f(v);
        if (res) v += res[(long)m * ldr + n];
        vv[r] = v;
        if (Cfz) Cfz[(long)m * ldc + n] = v;
        if (Cbz) Cbz[(long)m * ldc + n] = (__bf16)v;
      }
      if (Cbtz) {  // transposed store: contiguous 8 bf16 per lane along m
        bf16x8 pk;
#pragma unroll
        for (int r = 0; r < 8; ++r) pk[r] = (__bf16)vv[r];
        *(bf16x8*)(Cbtz + (long)n * ldc + m0 + wr * 32 + ms * 16 + hi * 8) = pk;
      }
    }
  }
}

// ---------------------------------------------------------------------------
// Flash attention: grid (S/64, H, B), 128 threads = 4 waves; each wave owns a
// 16-row strip. Scores via WMMA from global Q/K fragments; online softmax with
// 16-lane shfl_xor reductions; P staged bf16 in per-wave LDS (C->A layout);
// P*V via WMMA from pre-transposed Vt[h][dk][B*S]. Causal mask on diag tile.
// ---------------------------------------------------------------------------
__global__ void attn_flash(const __bf16* __restrict__ Q, const __bf16* __restrict__ Kc,
                           const __bf16* __restrict__ Vt, __bf16* __restrict__ Oc) {
  const int i0 = blockIdx.x * 64;
  const int h = blockIdx.y;
  const int b = blockIdx.z;
  const int tid = threadIdx.x;
  const int w = tid >> 5, lane = tid & 31;
  const int hi = lane >> 4, nl = lane & 15;

  const __bf16* Qh = Q + (long)h * MM * DKk;
  const __bf16* Kh = Kc + (long)h * MM * DKk;
  const __bf16* Vh = Vt + (long)h * DKk * MM;

  const long rowA = (long)b * SS + i0 + w * 16;
  bf16x16 qa0 = load_frag_a(Qh + rowA * DKk + 0, DKk);
  bf16x16 qa1 = load_frag_a(Qh + rowA * DKk + 32, DKk);

  __shared__ __bf16 pbuf[4][16 * 72];
  __bf16* pw = pbuf[w];

  f32x8 o[4];
#pragma unroll
  for (int s = 0; s < 4; ++s) o[s] = zero8();
  float Mr[8], Sr[8];
#pragma unroll
  for (int r = 0; r < 8; ++r) { Mr[r] = -3.0e38f; Sr[r] = 0.0f; }

  for (int t0 = 0; t0 <= i0; t0 += 64) {
    f32x8 sc[4];
#pragma unroll
    for (int sub = 0; sub < 4; ++sub) {
      long rowB = (long)b * SS + t0 + sub * 16;
      bf16x16 kb0 = load_frag_b(Kh + rowB * DKk + 0, DKk);
      bf16x16 kb1 = load_frag_b(Kh + rowB * DKk + 32, DKk);
      f32x8 c = zero8();
      c = wmma_bf16(qa0, kb0, c);
      c = wmma_bf16(qa1, kb1, c);
      sc[sub] = c * 0.125f;   // 1/sqrt(64)
    }
    if (t0 == i0) {  // causal mask on diagonal tile
#pragma unroll
      for (int sub = 0; sub < 4; ++sub)
#pragma unroll
        for (int r = 0; r < 8; ++r) {
          int t = t0 + sub * 16 + nl;
          int s = i0 + w * 16 + hi * 8 + r;
          if (t > s) sc[sub][r] = -3.0e38f;
        }
    }
    // online softmax per row (rows live in 16-lane halves)
#pragma unroll
    for (int r = 0; r < 8; ++r) {
      float v = fmaxf(fmaxf(sc[0][r], sc[1][r]), fmaxf(sc[2][r], sc[3][r]));
      v = fmaxf(v, __shfl_xor(v, 1));
      v = fmaxf(v, __shfl_xor(v, 2));
      v = fmaxf(v, __shfl_xor(v, 4));
      v = fmaxf(v, __shfl_xor(v, 8));
      float mnew = fmaxf(Mr[r], v);
      float corr = __expf(Mr[r] - mnew);
      Mr[r] = mnew;
      float ps = 0.0f;
#pragma unroll
      for (int sub = 0; sub < 4; ++sub) {
        float p = __expf(sc[sub][r] - mnew);
        pw[(hi * 8 + r) * 72 + sub * 16 + nl] = (__bf16)p;
        ps += p;
      }
      ps += __shfl_xor(ps, 1);
      ps += __shfl_xor(ps, 2);
      ps += __shfl_xor(ps, 4);
      ps += __shfl_xor(ps, 8);
      Sr[r] = Sr[r] * corr + ps;
#pragma unroll
      for (int sub = 0; sub < 4; ++sub) o[sub][r] *= corr;
    }
    __syncthreads();
    bf16x16 pa0 = load_frag_a(pw + 0, 72);
    bf16x16 pa1 = load_frag_a(pw + 32, 72);
#pragma unroll
    for (int sub = 0; sub < 4; ++sub) {
      const __bf16* vb = Vh + (long)(sub * 16) * MM + (long)b * SS + t0;
      bf16x16 v0 = load_frag_b(vb + 0, MM);
      bf16x16 v1 = load_frag_b(vb + 32, MM);
      o[sub] = wmma_bf16(pa0, v0, o[sub]);
      o[sub] = wmma_bf16(pa1, v1, o[sub]);
    }
    __syncthreads();
  }

  float inv[8];
#pragma unroll
  for (int r = 0; r < 8; ++r) inv[r] = 1.0f / Sr[r];
  long orow = (long)b * SS + i0 + w * 16 + hi * 8;
#pragma unroll
  for (int sub = 0; sub < 4; ++sub)
#pragma unroll
    for (int r = 0; r < 8; ++r)
      Oc[(orow + r) * DD + h * DKk + sub * 16 + nl] = (__bf16)(o[sub][r] * inv[r]);
}

// ---------------------------------------------------------------------------
extern "C" void kernel_launch(void* const* d_in, const int* in_sizes, int n_in,
                              void* d_out, int out_size, void* d_ws, size_t ws_size,
                              hipStream_t stream) {
  (void)in_sizes; (void)n_in; (void)out_size; (void)ws_size;
  const float* x     = (const float*)d_in[0];
  const float* Wq    = (const float*)d_in[1];
  const float* Wk    = (const float*)d_in[2];
  const float* Wv    = (const float*)d_in[3];
  const float* Wo    = (const float*)d_in[4];
  const float* W1    = (const float*)d_in[5];
  const float* b1    = (const float*)d_in[6];
  const float* W2    = (const float*)d_in[7];
  const float* b2    = (const float*)d_in[8];
  const float* g1    = (const float*)d_in[9];
  const float* bl1   = (const float*)d_in[10];
  const float* g2    = (const float*)d_in[11];
  const float* bl2   = (const float*)d_in[12];
  float* out = (float*)d_out;

  size_t off = 0;
  auto take = [&](size_t bytes) -> void* {
    void* p = (char*)d_ws + off;
    off += (bytes + 255) & ~(size_t)255;
    return p;
  };
  __bf16* h1   = (__bf16*)take((size_t)MM * DD * 2);
  __bf16* Wqt  = (__bf16*)take((size_t)HH * DD * DKk * 2);
  __bf16* Wkt  = (__bf16*)take((size_t)HH * DD * DKk * 2);
  __bf16* Wvt  = (__bf16*)take((size_t)HH * DD * DKk * 2);
  __bf16* Wot  = (__bf16*)take((size_t)DD * DD * 2);
  __bf16* W1t  = (__bf16*)take((size_t)DD * HID * 2);
  __bf16* W2t  = (__bf16*)take((size_t)HID * DD * 2);
  __bf16* Qb   = (__bf16*)take((size_t)HH * MM * DKk * 2);
  __bf16* Kb   = (__bf16*)take((size_t)HH * MM * DKk * 2);
  __bf16* Vtb  = (__bf16*)take((size_t)HH * DKk * MM * 2);
  __bf16* attn = (__bf16*)take((size_t)MM * DD * 2);
  float*  y    = (float*)take((size_t)MM * DD * 4);
  __bf16* h2   = (__bf16*)take((size_t)MM * DD * 2);
  __bf16* ub   = (__bf16*)take((size_t)MM * HID * 2);

  // weight transposes (fp32 -> bf16, [K][N] -> [N][K])
  transpose_cvt<<<dim3(DD / 64, DKk / 64, HH), 256, 0, stream>>>(Wq, Wqt, DD, DKk);
  transpose_cvt<<<dim3(DD / 64, DKk / 64, HH), 256, 0, stream>>>(Wk, Wkt, DD, DKk);
  transpose_cvt<<<dim3(DD / 64, DKk / 64, HH), 256, 0, stream>>>(Wv, Wvt, DD, DKk);
  transpose_cvt<<<dim3(DD / 64, DD / 64, 1), 256, 0, stream>>>(Wo, Wot, DD, DD);
  transpose_cvt<<<dim3(DD / 64, HID / 64, 1), 256, 0, stream>>>(W1, W1t, DD, HID);
  transpose_cvt<<<dim3(HID / 64, DD / 64, 1), 256, 0, stream>>>(W2, W2t, HID, DD);

  // LN1
  ln_bf16<<<MM, 256, 0, stream>>>(x, g1, bl1, h1, DD);

  // Q, K projections (per-head z batch); V stored transposed for P*V fragments
  gemm_tc<<<dim3(1, MM / 128, HH), 256, 0, stream>>>(
      h1, 0, DD, Wqt, (long)DKk * DD, DD,
      nullptr, Qb, nullptr, (long)MM * DKk, DKk,
      nullptr, nullptr, 0, DD, 0);
  gemm_tc<<<dim3(1, MM / 128, HH), 256, 0, stream>>>(
      h1, 0, DD, Wkt, (long)DKk * DD, DD,
      nullptr, Kb, nullptr, (long)MM * DKk, DKk,
      nullptr, nullptr, 0, DD, 0);
  gemm_tc<<<dim3(1, MM / 128, HH), 256, 0, stream>>>(
      h1, 0, DD, Wvt, (long)DKk * DD, DD,
      nullptr, nullptr, Vtb, (long)DKk * MM, MM,
      nullptr, nullptr, 0, DD, 0);

  // flash attention -> concat-head bf16 [M][768]
  attn_flash<<<dim3(SS / 64, HH, BB), 128, 0, stream>>>(Qb, Kb, Vtb, attn);

  // Wo projection + residual x  -> y (fp32)
  gemm_tc<<<dim3(DD / 64, MM / 128, 1), 256, 0, stream>>>(
      attn, 0, DD, Wot, 0, DD,
      y, nullptr, nullptr, 0, DD,
      nullptr, x, DD, DD, 0);

  // LN2
  ln_bf16<<<MM, 256, 0, stream>>>(y, g2, bl2, h2, DD);

  // MLP up + GELU -> u (bf16)
  gemm_tc<<<dim3(HID / 64, MM / 128, 1), 256, 0, stream>>>(
      h2, 0, DD, W1t, 0, DD,
      nullptr, ub, nullptr, 0, HID,
      b1, nullptr, 0, DD, 1);

  // MLP down + bias + residual y -> d_out (fp32)
  gemm_tc<<<dim3(DD / 64, MM / 128, 1), 256, 0, stream>>>(
      ub, 0, HID, W2t, 0, HID,
      out, nullptr, nullptr, 0, DD,
      b2, y, DD, HID, 0);
}
